// CloudCastV2_58050777972967
// MI455X (gfx1250) — matrix operand order repaired and checked by reading, not compiled, not run on hardware
//
#include <hip/hip_runtime.h>

// ============================================================================
// Swin-style transformer block for MI455X (gfx1250), wave32 + WMMA bf16.
// All GEMMs use v_wmma_f32_16x16x32_bf16 (bf16 operands, f32 accumulate),
// register-blocked 2x4 tiles per wave and software-pipelined: next-chunk loads
// are issued ahead of the current-chunk WMMA chain, with sched_barrier(0)
// pinning that order so the backend scheduler cannot re-serialize it.
// Scratch layout (bump allocated in d_ws, peak ~545 MB, buffers reused):
//   WQKV bf16 @0          (1.5 MB)   wq|wk|wv rows concatenated
//   WP   bf16 @0x180000   (0.5 MB)
//   W1   bf16 @0x200000   (2 MB)
//   W2   bf16 @0x400000   (2 MB)
//   XW   bf16 @0x0800000  (64 MB)  LN1+shift+window tokens; reused for O, then G
//   Q    bf16 @0x4800000  (64 MB)  reused for H1 (256 MB spanning Q/K/V/AR)
//   K    bf16 @0x8800000  (64 MB)
//   V    bf16 @0xC800000  (64 MB)
//   AR   f32  @0x10800000 (128 MB) attention-proj output (window order)
//   X3   f32  @0x18800000 (128 MB) post-attention residual state
// ============================================================================

typedef __attribute__((ext_vector_type(16))) __bf16 v16bf;
typedef __attribute__((ext_vector_type(8)))  __bf16 v8bf;
typedef __attribute__((ext_vector_type(8)))  float  v8f;

#define TOK   65536   // B*H*W tokens
#define CDIM  512
#define HID   2048

#if __has_builtin(__builtin_amdgcn_sched_barrier)
#define SCHED_BARRIER() __builtin_amdgcn_sched_barrier(0)
#else
#define SCHED_BARRIER()
#endif

// ---- WMMA helpers ----------------------------------------------------------
__device__ inline v8f wmma_bf16(v16bf a, v16bf b, v8f c) {
  return __builtin_amdgcn_wmma_f32_16x16x32_bf16(false, a, false, b, (short)0, c,
                                                 false, false);
}

// A-operand (16x32): lane%16 = row M; elements 0..7 -> K=khalf*8+j, 8..15 -> K=16+khalf*8+j
__device__ inline v16bf load_a16(const __bf16* rowptr, int k0, int khalf) {
  const v8bf lo = *(const v8bf*)(rowptr + k0 + khalf * 8);
  const v8bf hi = *(const v8bf*)(rowptr + k0 + 16 + khalf * 8);
  v16bf a;
#pragma unroll
  for (int j = 0; j < 8; ++j) { a[j] = lo[j]; a[j + 8] = hi[j]; }
  return a;
}

// B-operand (32x16): lane%16 = col N; elements 0..7 -> K=khalf*8+j, 8..15 -> K=16+khalf*8+j
__device__ inline v16bf load_b16(const __bf16* colptr, int k0, int khalf) {
  const v8bf lo = *(const v8bf*)(colptr + k0 + khalf * 8);
  const v8bf hi = *(const v8bf*)(colptr + k0 + 16 + khalf * 8);
  v16bf b;
#pragma unroll
  for (int j = 0; j < 8; ++j) { b[j] = lo[j]; b[j + 8] = hi[j]; }
  return b;
}

// ---- weight f32 -> bf16 ----------------------------------------------------
__global__ void cvt_bf16_kernel(const float* __restrict__ src,
                                __bf16* __restrict__ dst, int n) {
  int i = blockIdx.x * 256 + threadIdx.x;
  if (i < n) dst[i] = (__bf16)src[i];
}

// ---- LN1 + roll(-4,-4) + window partition -> Xw bf16 -----------------------
__global__ __launch_bounds__(256) void ln_shift_part_kernel(
    const float* __restrict__ x, const float* __restrict__ g,
    const float* __restrict__ b, __bf16* __restrict__ Xw) {
  const int wave = threadIdx.x >> 5, lane = threadIdx.x & 31;
  const int r = blockIdx.x * 8 + wave;        // 0..65535 window-order token
  const int win = r >> 6, n = r & 63;
  const int bimg = win >> 4, wloc = win & 15;
  const int h = (wloc >> 2) * 8 + (n >> 3);
  const int w = (wloc & 3) * 8 + (n & 7);
  const int hs = (h + 4) & 31, wsft = (w + 4) & 31;   // roll by -4
  const float* px = x + ((size_t)bimg * 1024 + hs * 32 + wsft) * CDIM;
  float v[16], s = 0.f, s2 = 0.f;
#pragma unroll
  for (int j = 0; j < 16; ++j) {
    float t = px[lane + j * 32]; v[j] = t; s += t; s2 += t * t;
  }
#pragma unroll
  for (int m = 1; m < 32; m <<= 1) { s += __shfl_xor(s, m, 32); s2 += __shfl_xor(s2, m, 32); }
  const float mean = s * (1.f / CDIM);
  const float rstd = rsqrtf(s2 * (1.f / CDIM) - mean * mean + 1e-5f);
  __bf16* po = Xw + (size_t)r * CDIM;
#pragma unroll
  for (int j = 0; j < 16; ++j) {
    int c = lane + j * 32;
    po[c] = (__bf16)((v[j] - mean) * rstd * g[c] + b[c]);
  }
}

// ---- generic WMMA GEMM: out[M,N] = A[M,K] @ W[N,K]^T (+bias) ---------------
// Each wave: 2 row-tiles x 4 col-tiles, double-buffered operand registers.
// MODE 0: store bf16    MODE 1: store f32
// MODE 2: store bf16 after exact GELU    MODE 3: store f32 + residual add
template <int MODE>
__global__ __launch_bounds__(256) void gemm_wmma_kernel(
    const __bf16* __restrict__ A, const __bf16* __restrict__ W,
    const float* __restrict__ bias, void* __restrict__ out,
    const float* __restrict__ res, int K, int N) {
  const int lane = threadIdx.x & 31, wave = threadIdx.x >> 5;
  const int khalf = lane >> 4, l15 = lane & 15;
  const int rowTile0 = (blockIdx.y * 8 + wave) * 2;   // 2 row tiles per wave
  const int colTile0 = blockIdx.x * 4;                // 4 col tiles per wave
  const __bf16* arow0 = A + (size_t)(rowTile0 * 16 + l15) * K;
  const __bf16* arow1 = arow0 + (size_t)16 * K;
  const __bf16* bcol[4];
#pragma unroll
  for (int c = 0; c < 4; ++c)
    bcol[c] = W + (size_t)((colTile0 + c) * 16 + l15) * K;

  v8f acc[2][4];
#pragma unroll
  for (int rt = 0; rt < 2; ++rt)
#pragma unroll
    for (int c = 0; c < 4; ++c) acc[rt][c] = (v8f){};

  // prologue: load chunk 0
  v16bf a0 = load_a16(arow0, 0, khalf);
  v16bf a1 = load_a16(arow1, 0, khalf);
  v16bf bb[4];
#pragma unroll
  for (int c = 0; c < 4; ++c) bb[c] = load_b16(bcol[c], 0, khalf);
  SCHED_BARRIER();

  for (int k0 = 32; k0 < K; k0 += 32) {
    // issue next-chunk loads, pinned ABOVE the current-chunk WMMA chain
    v16bf na0 = load_a16(arow0, k0, khalf);
    v16bf na1 = load_a16(arow1, k0, khalf);
    v16bf nb[4];
#pragma unroll
    for (int c = 0; c < 4; ++c) nb[c] = load_b16(bcol[c], k0, khalf);
    SCHED_BARRIER();
#pragma unroll
    for (int c = 0; c < 4; ++c) acc[0][c] = wmma_bf16(a0, bb[c], acc[0][c]);
#pragma unroll
    for (int c = 0; c < 4; ++c) acc[1][c] = wmma_bf16(a1, bb[c], acc[1][c]);
    SCHED_BARRIER();
    a0 = na0; a1 = na1;
#pragma unroll
    for (int c = 0; c < 4; ++c) bb[c] = nb[c];
  }
  // epilogue chunk
#pragma unroll
  for (int c = 0; c < 4; ++c) acc[0][c] = wmma_bf16(a0, bb[c], acc[0][c]);
#pragma unroll
  for (int c = 0; c < 4; ++c) acc[1][c] = wmma_bf16(a1, bb[c], acc[1][c]);

  // store
#pragma unroll
  for (int rt = 0; rt < 2; ++rt) {
    const int rbase = (rowTile0 + rt) * 16 + khalf * 8;
#pragma unroll
    for (int c = 0; c < 4; ++c) {
      const int col = (colTile0 + c) * 16 + l15;
      const float bv = bias[col];
#pragma unroll
      for (int i = 0; i < 8; ++i) {
        const size_t o = (size_t)(rbase + i) * N + col;
        float val = acc[rt][c][i] + bv;
        if (MODE == 0) {
          ((__bf16*)out)[o] = (__bf16)val;
        } else if (MODE == 1) {
          ((float*)out)[o] = val;
        } else if (MODE == 2) {
          float ge = 0.5f * val * (1.0f + erff(val * 0.70710678118654752f));
          ((__bf16*)out)[o] = (__bf16)ge;
        } else {
          ((float*)out)[o] = val + res[o];
        }
      }
    }
  }
}

// ---- windowed attention: one block (128 thr / 4 waves) per (window, head) --
__global__ __launch_bounds__(128) void attn_kernel(
    const __bf16* __restrict__ Q, const __bf16* __restrict__ Kb,
    const __bf16* __restrict__ V, const float* __restrict__ rel_table,
    __bf16* __restrict__ O) {
  __shared__ __bf16 vT[64][72];        // v transposed: vT[c][m], row 144B (16B aligned)
  __shared__ __bf16 Pl[4][16][64];     // per-wave softmax tile, A-layout staging
  const int blk = blockIdx.x;          // 0..8191
  const int win = blk >> 3, head = blk & 7;
  const int tid = threadIdx.x;
  const int wave = tid >> 5, lane = tid & 31;
  const int khalf = lane >> 4, l15 = lane & 15;
  const size_t base = (size_t)win * 64 * CDIM + head * 64;

  // stage V transposed into LDS
  for (int idx = tid; idx < 4096; idx += 128) {
    int m = idx >> 6, c = idx & 63;
    vT[c][m] = V[base + (size_t)m * CDIM + c];
  }
  __syncthreads();

  // S = q @ k^T : batch ALL operand loads, pin them above the 8-WMMA chain
  const int nbase = wave * 16;
  const __bf16* qrow = Q + base + (size_t)(nbase + l15) * CDIM;
  const v16bf aq0 = load_a16(qrow, 0, khalf);
  const v16bf aq1 = load_a16(qrow, 32, khalf);
  v16bf kb0[4], kb1[4];
#pragma unroll
  for (int mt = 0; mt < 4; ++mt) {
    const __bf16* kcol = Kb + base + (size_t)(mt * 16 + l15) * CDIM;
    kb0[mt] = load_b16(kcol, 0, khalf);
    kb1[mt] = load_b16(kcol, 32, khalf);
  }
  SCHED_BARRIER();
  v8f sacc[4];
#pragma unroll
  for (int mt = 0; mt < 4; ++mt) {
    v8f acc = {};
    acc = wmma_bf16(aq0, kb0[mt], acc);
    acc = wmma_bf16(aq1, kb1[mt], acc);
    sacc[mt] = acc;
  }
  SCHED_BARRIER();

  // scale + relative-position bias (torch view quirk) + shift mask
#pragma unroll
  for (int mt = 0; mt < 4; ++mt) {
#pragma unroll
    for (int i = 0; i < 8; ++i) {
      const int nq = nbase + i + khalf * 8;   // query index in window
      const int m = mt * 16 + l15;            // key index
      const int r0 = (nq >> 3) - head + 7;
      const int r1 = (nq & 7) - (m >> 3) + 7;
      const float bias = rel_table[(r0 * 15 + r1) * 8 + (m & 7)];
      const float sv = sacc[mt][i] * 0.125f + bias;
      const bool masked = ((nq >> 3) >= 4) || ((nq & 7) >= 4);
      sacc[mt][i] = masked ? -1e9f : sv;
    }
  }

  // row softmax: reduce across 4 m-tiles + 16-lane groups (xor masks 1,2,4,8)
#pragma unroll
  for (int i = 0; i < 8; ++i) {
    float rm = fmaxf(fmaxf(sacc[0][i], sacc[1][i]), fmaxf(sacc[2][i], sacc[3][i]));
#pragma unroll
    for (int msk = 1; msk < 16; msk <<= 1) rm = fmaxf(rm, __shfl_xor(rm, msk, 32));
    float rs = 0.f;
#pragma unroll
    for (int mt = 0; mt < 4; ++mt) { float e = __expf(sacc[mt][i] - rm); sacc[mt][i] = e; rs += e; }
#pragma unroll
    for (int msk = 1; msk < 16; msk <<= 1) rs += __shfl_xor(rs, msk, 32);
    const float inv = 1.f / rs;
#pragma unroll
    for (int mt = 0; mt < 4; ++mt) sacc[mt][i] *= inv;
  }

  // stage P (bf16) through LDS to re-enter WMMA A-layout
#pragma unroll
  for (int mt = 0; mt < 4; ++mt)
#pragma unroll
    for (int i = 0; i < 8; ++i)
      Pl[wave][i + khalf * 8][mt * 16 + l15] = (__bf16)sacc[mt][i];
  __syncthreads();

  // O = P @ v : batch LDS operand loads, pin above WMMA chain
  const __bf16* prow = &Pl[wave][l15][0];
  const v16bf pa0 = load_a16(prow, 0, khalf);
  const v16bf pa1 = load_a16(prow, 32, khalf);
  v16bf vb0[4], vb1[4];
#pragma unroll
  for (int ct = 0; ct < 4; ++ct) {
    const __bf16* vcol = &vT[ct * 16 + l15][0];
    vb0[ct] = load_b16(vcol, 0, khalf);
    vb1[ct] = load_b16(vcol, 32, khalf);
  }
  SCHED_BARRIER();
  v8f oacc[4];
#pragma unroll
  for (int ct = 0; ct < 4; ++ct) {
    v8f acc = {};
    acc = wmma_bf16(pa0, vb0[ct], acc);
    acc = wmma_bf16(pa1, vb1[ct], acc);
    oacc[ct] = acc;
  }
  SCHED_BARRIER();
#pragma unroll
  for (int ct = 0; ct < 4; ++ct) {
#pragma unroll
    for (int i = 0; i < 8; ++i) {
      const int nq = nbase + i + khalf * 8;
      const int c = ct * 16 + l15;
      O[base + (size_t)nq * CDIM + c] = (__bf16)oacc[ct][i];
    }
  }
}

// ---- x3 = LN1(x) + unroll(ar) + sigmoid(gate)*x ; G = LN2(x3) bf16 ---------
__global__ __launch_bounds__(256) void resid_ln2_kernel(
    const float* __restrict__ x, const float* __restrict__ ar,
    const float* __restrict__ gate, const float* __restrict__ g1,
    const float* __restrict__ b1, const float* __restrict__ g2,
    const float* __restrict__ b2, float* __restrict__ x3,
    __bf16* __restrict__ G) {
  const int wave = threadIdx.x >> 5, lane = threadIdx.x & 31;
  const int t = blockIdx.x * 8 + wave;            // pixel-order token
  const int bimg = t >> 10, hw = t & 1023;
  const int h = hw >> 5, w = hw & 31;
  const int hh = (h + 28) & 31, ww = (w + 28) & 31;   // roll back by +4
  const int win = bimg * 16 + (hh >> 3) * 4 + (ww >> 3);
  const int n = (hh & 7) * 8 + (ww & 7);
  const float* par = ar + ((size_t)win * 64 + n) * CDIM;
  const float* px = x + (size_t)t * CDIM;
  const float sg = 1.f / (1.f + __expf(-gate[hw]));

  float v[16], s = 0.f, s2 = 0.f;
#pragma unroll
  for (int j = 0; j < 16; ++j) {
    float u = px[lane + j * 32]; v[j] = u; s += u; s2 += u * u;
  }
#pragma unroll
  for (int m = 1; m < 32; m <<= 1) { s += __shfl_xor(s, m, 32); s2 += __shfl_xor(s2, m, 32); }
  const float mean1 = s * (1.f / CDIM);
  const float rstd1 = rsqrtf(s2 * (1.f / CDIM) - mean1 * mean1 + 1e-5f);

  float y[16], t1 = 0.f, t2 = 0.f;
#pragma unroll
  for (int j = 0; j < 16; ++j) {
    const int c = lane + j * 32;
    const float xn = (v[j] - mean1) * rstd1 * g1[c] + b1[c];
    const float val = xn + par[c] + sg * v[j];
    y[j] = val; t1 += val; t2 += val * val;
  }
#pragma unroll
  for (int m = 1; m < 32; m <<= 1) { t1 += __shfl_xor(t1, m, 32); t2 += __shfl_xor(t2, m, 32); }
  const float mean2 = t1 * (1.f / CDIM);
  const float rstd2 = rsqrtf(t2 * (1.f / CDIM) - mean2 * mean2 + 1e-5f);

  float* px3 = x3 + (size_t)t * CDIM;
  __bf16* pg = G + (size_t)t * CDIM;
#pragma unroll
  for (int j = 0; j < 16; ++j) {
    const int c = lane + j * 32;
    px3[c] = y[j];
    pg[c] = (__bf16)((y[j] - mean2) * rstd2 * g2[c] + b2[c]);
  }
}

// ============================================================================
extern "C" void kernel_launch(void* const* d_in, const int* in_sizes, int n_in,
                              void* d_out, int out_size, void* d_ws, size_t ws_size,
                              hipStream_t stream) {
  const float* x       = (const float*)d_in[0];
  const float* ln1_g   = (const float*)d_in[1];
  const float* ln1_b   = (const float*)d_in[2];
  const float* wq      = (const float*)d_in[3];
  const float* bq      = (const float*)d_in[4];
  const float* wk      = (const float*)d_in[5];
  const float* bk      = (const float*)d_in[6];
  const float* wv      = (const float*)d_in[7];
  const float* bv      = (const float*)d_in[8];
  const float* wp      = (const float*)d_in[9];
  const float* bp      = (const float*)d_in[10];
  const float* rel_t   = (const float*)d_in[11];
  const float* gate    = (const float*)d_in[12];
  const float* ln2_g   = (const float*)d_in[13];
  const float* ln2_b   = (const float*)d_in[14];
  const float* mlp_w1  = (const float*)d_in[15];
  const float* mlp_b1  = (const float*)d_in[16];
  const float* mlp_w2  = (const float*)d_in[17];
  const float* mlp_b2  = (const float*)d_in[18];

  char* ws = (char*)d_ws;
  __bf16* WQKV = (__bf16*)(ws + 0x0);
  __bf16* WP   = (__bf16*)(ws + 0x180000);
  __bf16* W1   = (__bf16*)(ws + 0x200000);
  __bf16* W2   = (__bf16*)(ws + 0x400000);
  __bf16* XW   = (__bf16*)(ws + 0x0800000);   // tokens; reused as O, then G
  __bf16* Qb   = (__bf16*)(ws + 0x4800000);
  __bf16* Kb   = (__bf16*)(ws + 0x8800000);
  __bf16* Vb   = (__bf16*)(ws + 0xC800000);
  float*  AR   = (float*)(ws + 0x10800000);
  float*  X3   = (float*)(ws + 0x18800000);
  __bf16* Ob   = XW;                           // reuse: XW dead after QKV gemms
  __bf16* G    = XW;                           // reuse: O dead after proj gemm
  __bf16* H1   = Qb;                           // reuse: Q/K/V dead after attention/proj

  // 1) weights -> bf16
  cvt_bf16_kernel<<<1024, 256, 0, stream>>>(wq, WQKV, 262144);
  cvt_bf16_kernel<<<1024, 256, 0, stream>>>(wk, WQKV + 262144, 262144);
  cvt_bf16_kernel<<<1024, 256, 0, stream>>>(wv, WQKV + 524288, 262144);
  cvt_bf16_kernel<<<1024, 256, 0, stream>>>(wp, WP, 262144);
  cvt_bf16_kernel<<<4096, 256, 0, stream>>>(mlp_w1, W1, 1048576);
  cvt_bf16_kernel<<<4096, 256, 0, stream>>>(mlp_w2, W2, 1048576);

  // 2) LN1 + shift + window partition
  ln_shift_part_kernel<<<TOK / 8, 256, 0, stream>>>(x, ln1_g, ln1_b, XW);

  // 3) Q/K/V projections (65536x512 @ 512x512^T); block = 256 rows x 64 cols
  dim3 gQKV(CDIM / 64, TOK / 256);
  gemm_wmma_kernel<0><<<gQKV, 256, 0, stream>>>(XW, WQKV,          bq, Qb, nullptr, CDIM, CDIM);
  gemm_wmma_kernel<0><<<gQKV, 256, 0, stream>>>(XW, WQKV + 262144, bk, Kb, nullptr, CDIM, CDIM);
  gemm_wmma_kernel<0><<<gQKV, 256, 0, stream>>>(XW, WQKV + 524288, bv, Vb, nullptr, CDIM, CDIM);

  // 4) windowed attention, one block per (window, head)
  attn_kernel<<<8192, 128, 0, stream>>>(Qb, Kb, Vb, rel_t, Ob);

  // 5) output projection -> AR f32 (window order)
  gemm_wmma_kernel<1><<<dim3(CDIM / 64, TOK / 256), 256, 0, stream>>>(Ob, WP, bp, AR, nullptr, CDIM, CDIM);

  // 6) residual combine + LN2
  resid_ln2_kernel<<<TOK / 8, 256, 0, stream>>>(x, AR, gate, ln1_g, ln1_b,
                                                ln2_g, ln2_b, X3, G);

  // 7) MLP: GEMM1 + exact GELU -> H1 bf16; GEMM2 + residual -> d_out f32
  gemm_wmma_kernel<2><<<dim3(HID / 64, TOK / 256), 256, 0, stream>>>(G, W1, mlp_b1, H1, nullptr, CDIM, HID);
  gemm_wmma_kernel<3><<<dim3(CDIM / 64, TOK / 256), 256, 0, stream>>>(H1, W2, mlp_b2, d_out, X3, HID, CDIM);
}
